// WebGraphEncoder_82918638616927
// MI455X (gfx1250) — matrix operand order repaired and checked by reading, not compiled, not run on hardware
//
#include <hip/hip_runtime.h>

typedef __attribute__((ext_vector_type(16))) _Float16 v16h;
typedef __attribute__((ext_vector_type(8)))  _Float16 v8h;
typedef __attribute__((ext_vector_type(8)))  float    v8f;

#define NNODES  25000
#define NEDGES  400000
#define IN_DIM  174
#define HID_DIM 128
#define OUT_DIM 64
#define K1PAD   192          // IN_DIM padded to multiple of 32
#define MTILES  1563         // ceil(25000/16)
#define MPAD    (MTILES*16)  // 25008

__device__ __forceinline__ void atomicAddF(float* p, float v) {
  // hardware global_atomic_add_f32, no-return (result discarded)
  (void)__hip_atomic_fetch_add(p, v, __ATOMIC_RELAXED, __HIP_MEMORY_SCOPE_AGENT);
}

// ---------------- utility kernels ----------------

__global__ void fill0_f32(float* __restrict__ p, long n) {
  long i = (long)blockIdx.x * blockDim.x + threadIdx.x;
  if (i < n) p[i] = 0.0f;
}

// convert f32 [rs,cs] -> f16 [rd,cd], zero padding
__global__ void cvt_pad_f16(const float* __restrict__ src, _Float16* __restrict__ dst,
                            int rs, int cs, int rd, int cd) {
  long i = (long)blockIdx.x * blockDim.x + threadIdx.x;
  long n = (long)rd * cd;
  if (i >= n) return;
  int r = (int)(i / cd), c = (int)(i % cd);
  float v = (r < rs && c < cs) ? src[(long)r * cs + c] : 0.0f;
  dst[i] = (_Float16)v;
}

// Pack two weight matrices Wa [Na,Kreal], Wb [Na..Ntot rows, Kreal] (row-major f32,
// PyG convention W[outF][inF]) into the WMMA B-operand tile layout:
// out[nt][kt][lane][e], N = nt*16 + lane%16, K = kt*32 + 16*(lane/16) + e.
__global__ void pack_b(const float* __restrict__ Wa, const float* __restrict__ Wb,
                       _Float16* __restrict__ out, int Na, int Ntot, int Kreal, int kTiles) {
  long i = (long)blockIdx.x * blockDim.x + threadIdx.x;
  long n = (long)(Ntot / 16) * kTiles * 512;
  if (i >= n) return;
  int e  = (int)(i & 15);
  int ln = (int)((i >> 4) & 31);
  long t = i >> 9;
  int kt = (int)(t % kTiles);
  int nt = (int)(t / kTiles);
  int N = nt * 16 + (ln & 15);
  int K = kt * 32 + ((ln >> 4) << 4) + e;
  float v = 0.0f;
  if (K < Kreal)
    v = (N < Na) ? Wa[(long)N * Kreal + K] : Wb[(long)(N - Na) * Kreal + K];
  out[i] = (_Float16)v;
}

__global__ void deg_count(const int* __restrict__ dst, float* __restrict__ deg) {
  int e = blockIdx.x * blockDim.x + threadIdx.x;
  if (e < NEDGES) atomicAddF(&deg[dst[e]], 1.0f);
}

// ---------------- WMMA GEMM ----------------
// C[MPAD, Ntot] = A[MPAD, Kpad](f16) @ Bp(packed tiles), one 16x16 tile per wave.
__global__ void __launch_bounds__(256) gemm_wmma(
    const _Float16* __restrict__ A, const _Float16* __restrict__ Bp,
    float* __restrict__ C, int Kpad, int Ntot) {
  int wave = threadIdx.x >> 5;
  int lane = threadIdx.x & 31;
  int mt = blockIdx.x * 8 + wave;      // wave-uniform
  int nt = blockIdx.y;
  if (mt >= MTILES) return;            // uniform per wave: EXEC stays all-ones
  int m = lane & 15;
  int g = lane >> 4;
  int kTiles = Kpad >> 5;
  // A operand, this code's contiguous k order (matches packed B): lane holds row M=lane%16,
  // elements 0..15 -> K = 16*g + e.
  const _Float16* arow = A + (size_t)(mt * 16 + m) * Kpad + 16 * g;
  const _Float16* bptr = Bp + (size_t)nt * kTiles * 512 + lane * 16;
  v8f acc = {};
  for (int kt = 0; kt < kTiles; ++kt) {
    v8h alo = *(const v8h*)(arow + kt * 32);
    v8h ahi = *(const v8h*)(arow + kt * 32 + 8);
    v16h a = __builtin_shufflevector(alo, ahi, 0,1,2,3,4,5,6,7,8,9,10,11,12,13,14,15);
    v16h b = *(const v16h*)(bptr + (size_t)kt * 512);
    acc = __builtin_amdgcn_wmma_f32_16x16x32_f16(
        /*neg_a=*/false, a, /*neg_b=*/false, b,
        /*c_mod=*/(short)0, acc, /*reuse_a=*/false, /*reuse_b=*/false);
  }
  // C/D f32 layout: N = lane%16, M = v + 8*(lane/16)
  int row0 = mt * 16 + 8 * g;
  int col  = nt * 16 + m;
#pragma unroll
  for (int v = 0; v < 8; ++v)
    C[(size_t)(row0 + v) * Ntot + col] = acc[v];
}

// ---------------- edge scatter-add ----------------
// S[dst, f] += G[src, colOff + f], f = 0..F-1, float4 granularity.
__global__ void scatter_add(const float* __restrict__ G, int ldg, int colOff,
                            const int* __restrict__ src, const int* __restrict__ dst,
                            float* __restrict__ S, int F) {
  long i = (long)blockIdx.x * blockDim.x + threadIdx.x;
  int chunks = F >> 2;
  long n = (long)NEDGES * chunks;
  if (i >= n) return;
  int e = (int)(i / chunks);
  int c = (int)(i % chunks);
  if (e + 64 < NEDGES)  // speculative prefetch of a future source row chunk
    __builtin_prefetch(G + (size_t)src[e + 64] * ldg + colOff + c * 4, 0, 0);
  int s = src[e], d = dst[e];
  float4 v = *(const float4*)(G + (size_t)s * ldg + colOff + c * 4);
  float* sp = S + (size_t)d * F + c * 4;
  atomicAddF(sp + 0, v.x);
  atomicAddF(sp + 1, v.y);
  atomicAddF(sp + 2, v.z);
  atomicAddF(sp + 3, v.w);
}

// ---------------- combine stages ----------------
// H1 = relu(S1/max(deg,1) + b1 + G1[:,128:256]) -> f16 [MPAD,128]
__global__ void combine1(const float* __restrict__ S, const float* __restrict__ G,
                         const float* __restrict__ b, const float* __restrict__ deg,
                         _Float16* __restrict__ H) {
  long i = (long)blockIdx.x * blockDim.x + threadIdx.x;
  long n = (long)MPAD * HID_DIM;
  if (i >= n) return;
  int r = (int)(i / HID_DIM), f = (int)(i % HID_DIM);
  float v = 0.0f;
  if (r < NNODES) {
    float d = fmaxf(deg[r], 1.0f);
    v = S[(size_t)r * HID_DIM + f] / d + b[f] + G[(size_t)r * 256 + HID_DIM + f];
    v = fmaxf(v, 0.0f);
  }
  H[i] = (_Float16)v;
}

// out = S2/max(deg,1) + b2 + G2[:,64:128]
__global__ void combine2(const float* __restrict__ S, const float* __restrict__ G,
                         const float* __restrict__ b, const float* __restrict__ deg,
                         float* __restrict__ out) {
  long i = (long)blockIdx.x * blockDim.x + threadIdx.x;
  long n = (long)NNODES * OUT_DIM;
  if (i >= n) return;
  int r = (int)(i / OUT_DIM), f = (int)(i % OUT_DIM);
  float d = fmaxf(deg[r], 1.0f);
  out[i] = S[(size_t)r * OUT_DIM + f] / d + b[f] + G[(size_t)r * 128 + OUT_DIM + f];
}

// ---------------- launcher ----------------

extern "C" void kernel_launch(void* const* d_in, const int* in_sizes, int n_in,
                              void* d_out, int out_size, void* d_ws, size_t ws_size,
                              hipStream_t stream) {
  (void)in_sizes; (void)n_in; (void)out_size; (void)ws_size;
  const float* x    = (const float*)d_in[0];
  const int*   edge = (const int*)d_in[1];
  const float* W1l  = (const float*)d_in[2];
  const float* b1   = (const float*)d_in[3];
  const float* W1r  = (const float*)d_in[4];
  const float* W2l  = (const float*)d_in[5];
  const float* b2   = (const float*)d_in[6];
  const float* W2r  = (const float*)d_in[7];
  const int* srcI = edge;
  const int* dstI = edge + NEDGES;
  float* out = (float*)d_out;

  char* ws = (char*)d_ws;
  size_t off = 0;
  auto alloc = [&](size_t bytes) -> void* {
    void* p = ws + off;
    off += (bytes + 511) & ~(size_t)511;
    return p;
  };
  float*    deg = (float*)   alloc((size_t)NNODES * 4);
  _Float16* Xh  = (_Float16*)alloc((size_t)MPAD * K1PAD * 2);
  _Float16* B1  = (_Float16*)alloc((size_t)16 * 6 * 512 * 2);   // Ntot=256, kTiles=6
  float*    G1  = (float*)   alloc((size_t)MPAD * 256 * 4);
  float*    S1  = (float*)   alloc((size_t)NNODES * HID_DIM * 4);
  _Float16* H1h = (_Float16*)alloc((size_t)MPAD * HID_DIM * 2);
  _Float16* B2  = (_Float16*)alloc((size_t)8 * 4 * 512 * 2);    // Ntot=128, kTiles=4
  float*    G2  = (float*)   alloc((size_t)MPAD * 128 * 4);
  float*    S2  = (float*)   alloc((size_t)NNODES * OUT_DIM * 4);

  const int T = 256;
  auto blocks = [](long n, int t) { return (int)((n + t - 1) / t); };

  // zero-init accumulators / degree
  fill0_f32<<<blocks((long)NNODES, T), T, 0, stream>>>(deg, NNODES);
  fill0_f32<<<blocks((long)NNODES * HID_DIM, T), T, 0, stream>>>(S1, (long)NNODES * HID_DIM);
  fill0_f32<<<blocks((long)NNODES * OUT_DIM, T), T, 0, stream>>>(S2, (long)NNODES * OUT_DIM);

  // degree
  deg_count<<<blocks(NEDGES, T), T, 0, stream>>>(dstI, deg);

  // f16 convert + pad of node features
  cvt_pad_f16<<<blocks((long)MPAD * K1PAD, T), T, 0, stream>>>(x, Xh, NNODES, IN_DIM, MPAD, K1PAD);

  // pack weights into WMMA B layout: layer1 cols [0:128)=W1_l, [128:256)=W1_r
  pack_b<<<blocks((long)16 * 6 * 512, T), T, 0, stream>>>(W1l, W1r, B1, HID_DIM, 256, IN_DIM, 6);
  // layer2 cols [0:64)=W2_l, [64:128)=W2_r
  pack_b<<<blocks((long)8 * 4 * 512, T), T, 0, stream>>>(W2l, W2r, B2, OUT_DIM, 128, HID_DIM, 4);

  // GEMM1: G1[MPAD,256] = Xh @ [W1_l;W1_r]^T
  {
    dim3 grid((MTILES + 7) / 8, 256 / 16);
    gemm_wmma<<<grid, T, 0, stream>>>(Xh, B1, G1, K1PAD, 256);
  }

  // scatter Y1 = G1[:, :128] over edges into S1
  scatter_add<<<blocks((long)NEDGES * (HID_DIM / 4), T), T, 0, stream>>>(
      G1, 256, 0, srcI, dstI, S1, HID_DIM);

  // H1 = relu(mean + b1 + root), emitted as f16 A-operand for layer 2
  combine1<<<blocks((long)MPAD * HID_DIM, T), T, 0, stream>>>(S1, G1, b1, deg, H1h);

  // GEMM2: G2[MPAD,128] = H1h @ [W2_l;W2_r]^T
  {
    dim3 grid((MTILES + 7) / 8, 128 / 16);
    gemm_wmma<<<grid, T, 0, stream>>>(H1h, B2, G2, HID_DIM, 128);
  }

  // scatter Y2 = G2[:, :64] into S2
  scatter_add<<<blocks((long)NEDGES * (OUT_DIM / 4), T), T, 0, stream>>>(
      G2, 128, 0, srcI, dstI, S2, OUT_DIM);

  // final output
  combine2<<<blocks((long)NNODES * OUT_DIM, T), T, 0, stream>>>(S2, G2, b2, deg, out);
}
